// GeneratorNetV2_10230612099727
// MI455X (gfx1250) — matrix operand order repaired and checked by reading, not compiled, not run on hardware
//
#include <hip/hip_runtime.h>
#include <hip/hip_bf16.h>
#include <stdint.h>

// ---------------- constants from the reference ----------------
#define BATCH 4096
#define ZDIM  128
#define NNODE 64
#define HOPS  256
#define NOPS  8
#define H1    512
#define H2    1024
#define H3    2048
#define NN2   4096            // NNODE*NNODE

#ifndef USE_TDM
#define USE_TDM 1             // flip to 0 if tensor_load_to_lds asm is rejected
#endif

typedef __attribute__((ext_vector_type(16))) __bf16 v16bf;
typedef __attribute__((ext_vector_type(8)))  float  v8f;
typedef __attribute__((ext_vector_type(4)))  unsigned int u32x4;
typedef __attribute__((ext_vector_type(8)))  unsigned int u32x8;
typedef int v4i __attribute__((vector_size(16)));   // matches builtin prototype spelling

__device__ __forceinline__ unsigned short f2bf(float f) {
  uint32_t u = __float_as_uint(f);
  u += 0x7FFFu + ((u >> 16) & 1u);          // round-to-nearest-even
  return (unsigned short)(u >> 16);
}

// ---------------- prep kernels (one-time weight transforms) ----------------
__global__ void k_cvt_bf16(const float* __restrict__ s, unsigned short* __restrict__ d, size_t n) {
  size_t i = (size_t)blockIdx.x * blockDim.x + threadIdx.x;
  if (i < n) d[i] = f2bf(s[i]);
}

// Tiled transpose + f32->bf16 convert: dst[c][r] = cvt(src[r][c])  (dst is [C][R])
// DIFF variant: dst[c][r] = cvt(src[r][NN2+c] - src[r][c])   (for We class-1 minus class-0)
// blockDim (32,8); R, C multiples of 32.
template<bool DIFF>
__global__ void k_transpose_cvt(const float* __restrict__ src, int srcLD,
                                unsigned short* __restrict__ dst, int R, int C) {
  __shared__ float t[32][33];
  const int r0 = blockIdx.y * 32, c0 = blockIdx.x * 32;
  const int tx = threadIdx.x, ty = threadIdx.y;
#pragma unroll
  for (int i = 0; i < 32; i += 8) {
    size_t r = r0 + ty + i, c = c0 + tx;
    float v = DIFF ? (src[r * srcLD + NN2 + c] - src[r * srcLD + c])
                   : src[r * srcLD + c];
    t[ty + i][tx] = v;
  }
  __syncthreads();
#pragma unroll
  for (int i = 0; i < 32; i += 8) {
    size_t c = c0 + ty + i, r = r0 + tx;
    dst[c * R + r] = f2bf(t[tx][ty + i]);
  }
}

// Wf_tmp[k][n*8+o] = sum_h Wn[k][n*256+h] * Wm[h][o]   (f32; transposed+converted later)
__global__ void k_prep_wf(const float* __restrict__ Wn, const float* __restrict__ Wm,
                          float* __restrict__ Wf) {
  size_t idx = (size_t)blockIdx.x * blockDim.x + threadIdx.x;
  if (idx >= (size_t)H3 * NNODE * NOPS) return;
  size_t k = idx >> 9;
  int col = (int)(idx & 511), node = col >> 3, o = col & 7;
  const float* wn = Wn + k * (NNODE * HOPS) + (size_t)node * HOPS;
  float acc = 0.f;
  for (int h = 0; h < HOPS; ++h) acc += wn[h] * Wm[h * NOPS + o];
  Wf[idx] = acc;
}

__global__ void k_prep_bd(const float* __restrict__ be, float* __restrict__ bd) {
  int ij = blockIdx.x * blockDim.x + threadIdx.x;
  if (ij < NN2) bd[ij] = be[NN2 + ij] - be[ij];
}

__global__ void k_prep_bf(const float* __restrict__ bn, const float* __restrict__ Wm,
                          float* __restrict__ bf) {
  int col = blockIdx.x * blockDim.x + threadIdx.x;
  if (col >= NNODE * NOPS) return;
  int node = col >> 3, o = col & 7;
  float acc = 0.f;
  for (int h = 0; h < HOPS; ++h) acc += bn[node * HOPS + h] * Wm[h * NOPS + o];
  bf[col] = acc;
}

// ---------------- probe: async global->LDS path (never launched) ----------------
#if __has_builtin(__builtin_amdgcn_global_load_async_to_lds_b128)
__global__ void k_probe_async(const unsigned short* __restrict__ g, unsigned short* __restrict__ o) {
  __shared__ unsigned short buf[2048];
  __attribute__((address_space(1))) v4i* gp =
      (__attribute__((address_space(1))) v4i*)(g + threadIdx.x * 8);
  __attribute__((address_space(3))) v4i* lp =
      (__attribute__((address_space(3))) v4i*)&buf[threadIdx.x * 8];
  __builtin_amdgcn_global_load_async_to_lds_b128(gp, lp, 0, 0);
#if __has_builtin(__builtin_amdgcn_s_wait_asynccnt)
  __builtin_amdgcn_s_wait_asynccnt(0);
#endif
  __syncthreads();
  o[threadIdx.x] = buf[(threadIdx.x * 8) ^ 8];
}
#endif

// ---------------- bf16 WMMA GEMM:  out = act(A[M,K] @ Wt[N,K]^T + bias) ----------------
// Weights pre-transposed: Wt is [N, K] row-major (so B-tile staging is address-preserving).
// 256 threads = 8 wave32; block tile 128(M) x 128(N); wave tile 32x64 (2x4 WMMA, 8/K-step).
template<bool TANH_BF16_OUT>
__global__ __launch_bounds__(256)
void k_gemm_bf16(const unsigned short* __restrict__ A,
                 const unsigned short* __restrict__ Wt,
                 const float* __restrict__ bias,
                 void* __restrict__ out,
                 int N, int K) {
  constexpr int BSTRIDE = 40;                 // 32 + 8 pad (TDM pad: 16DW interval, 4DW amount)
  __shared__ unsigned short Bs[128 * BSTRIDE];

  const int tid  = threadIdx.x;
  const int lane = tid & 31;
  const int wid  = tid >> 5;
  const int wm   = wid >> 1;                  // 4 wave rows (M)
  const int wn   = wid & 1;                   // 2 wave cols (N)
  const int m0   = blockIdx.y * 128 + wm * 32;
  const int nb   = blockIdx.x * 128;
  const int half = lane >> 4;                 // 0|1
  const int l15  = lane & 15;

  v8f acc[2][4] = {};

#if USE_TDM
  const unsigned int ldsBase =
      (unsigned int)(size_t)(__attribute__((address_space(3))) unsigned short*)Bs;
#endif

  for (int k0 = 0; k0 < K; k0 += 32) {
#if USE_TDM
    // ---- Tensor Data Mover: 2D tile [128 rows x 32 elems] of bf16, global -> LDS,
    //      LDS padding 4 DWORDs every 16 DWORDs => row stride 40 elements.
    if (wid == 0) {
      unsigned long long ga = (unsigned long long)(size_t)(Wt + (size_t)nb * K + k0);
      u32x4 g0;
      g0[0] = 1u;                                                    // count=1, user mode
      g0[1] = ldsBase;                                               // lds_addr
      g0[2] = (unsigned int)ga;                                      // global_addr[31:0]
      g0[3] = (unsigned int)((ga >> 32) & 0x1FFFFFFull) | (2u << 30); // addr[56:32], type=2
      u32x8 g1;
      g1[0] = (1u << 16)                                             // data_size = 2B
            | (1u << 20)                                             // pad_enable
            | (3u << 22)                                             // pad_interval = 16 DW
            | (3u << 25);                                            // pad_amount  = 4 DW
      g1[1] = ((unsigned int)K & 0xFFFFu) << 16;                     // tensor_dim0[15:0]
      g1[2] = ((unsigned int)K >> 16) | (((unsigned int)N & 0xFFFFu) << 16); // dim0 hi, dim1 lo
      g1[3] = ((unsigned int)N >> 16) | (32u << 16);                 // dim1 hi, tile_dim0=32
      g1[4] = 128u;                                                  // tile_dim1=128, tile_dim2=0
      g1[5] = (unsigned int)K;                                       // tensor_dim0_stride lo
      g1[6] = 0u;
      g1[7] = 0u;
      asm volatile("tensor_load_to_lds %0, %1" :: "s"(g0), "s"(g1) : "memory");
      __builtin_amdgcn_s_wait_tensorcnt(0);
    }
    __syncthreads();
#else
    // ---- manual staging: 512 chunks of 8 bf16, b128 load + b128 LDS store ----
#pragma unroll
    for (int s = 0; s < 2; ++s) {
      int c  = tid * 2 + s;
      int n  = c >> 2;
      int k8 = (c & 3) << 3;
      uint4 q = *(const uint4*)(Wt + (size_t)(nb + n) * K + k0 + k8);
      *(uint4*)&Bs[n * BSTRIDE + k8] = q;
    }
    __syncthreads();
#endif

    // ---- A fragments: 16x32 bf16 (half selects K 0-7/16-23 vs 8-15/24-31) ----
    v16bf afrag[2], bfrag[4];
#pragma unroll
    for (int sm = 0; sm < 2; ++sm) {
      const unsigned short* p0 = A + (size_t)(m0 + sm * 16 + l15) * K + k0 + half * 8;
      if (k0 + 32 < K) __builtin_prefetch(p0 + 32, 0, 3);
      uint32_t* w = (uint32_t*)&afrag[sm];
      const uint32_t* q0 = (const uint32_t*)p0;
      const uint32_t* q1 = (const uint32_t*)(p0 + 16);
      w[0] = q0[0]; w[1] = q0[1]; w[2] = q0[2]; w[3] = q0[3];
      w[4] = q1[0]; w[5] = q1[1]; w[6] = q1[2]; w[7] = q1[3];
    }
    // ---- B fragments: 32x16 bf16 (half selects K 0-7/16-23 vs 8-15/24-31, col = lane&15) ----
#pragma unroll
    for (int sn = 0; sn < 4; ++sn) {
      const unsigned short* p = &Bs[(wn * 64 + sn * 16 + l15) * BSTRIDE + half * 8];
      uint32_t* w = (uint32_t*)&bfrag[sn];
      const uint32_t* q = (const uint32_t*)p;
      w[0] = q[0]; w[1] = q[1]; w[2] = q[2]; w[3] = q[3];
      w[4] = q[8]; w[5] = q[9]; w[6] = q[10]; w[7] = q[11];
    }
#pragma unroll
    for (int sm = 0; sm < 2; ++sm)
#pragma unroll
      for (int sn = 0; sn < 4; ++sn)
        acc[sm][sn] = __builtin_amdgcn_wmma_f32_16x16x32_bf16(
            false, afrag[sm], false, bfrag[sn], (short)0, acc[sm][sn], false, false);
    __syncthreads();
  }

  // ---- epilogue: C layout VGPR r -> row r (+8 for upper lane half), col = lane&15 ----
#pragma unroll
  for (int sm = 0; sm < 2; ++sm)
#pragma unroll
    for (int sn = 0; sn < 4; ++sn) {
      int gcol = nb + wn * 64 + sn * 16 + l15;
      float bv = bias[gcol];
#pragma unroll
      for (int r = 0; r < 8; ++r) {
        int grow = m0 + sm * 16 + half * 8 + r;
        float v = acc[sm][sn][r] + bv;
        if (TANH_BF16_OUT)
          ((unsigned short*)out)[(size_t)grow * N + gcol] = f2bf(tanhf(v));
        else
          ((float*)out)[(size_t)grow * N + gcol] = v;
      }
    }
}

// ---------------- per-batch graph kernel ----------------
__global__ __launch_bounds__(256)
void k_graph(const float* __restrict__ dscore,   // [B, 64*64]  (s1-s0, bias included)
             const float* __restrict__ hproj,    // [B, 64*8]   (bn@Wm included)
             const float* __restrict__ g_edges,  // [B,64,64,2]
             const float* __restrict__ g_ops,    // [B,64,8]
             const float* __restrict__ bm,       // [8]
             float* __restrict__ out) {          // [B,64,8]
  __shared__ unsigned long long colmask[NNODE];
  __shared__ float dinv_s[NNODE];
  __shared__ float msg_s[NNODE][NOPS];
  __shared__ float emb_s[NNODE][NOPS];

  const int b = blockIdx.x, tid = threadIdx.x;
  if (tid < NNODE) colmask[tid] = 0ull;
  __syncthreads();

  const float* ds = dscore + (size_t)b * NN2;
  for (int p = tid; p < NN2; p += 256) {
    int i = p >> 6, j = p & 63;
    if (i < j) {
      float dsym = 0.5f * (ds[i * 64 + j] + ds[j * 64 + i]);
      const float* ge = g_edges + ((size_t)(b * NNODE + i) * NNODE + j) * 2;
      if (dsym + ge[1] > ge[0])                      // argmax==class1 (ties -> class0)
        atomicOr(&colmask[j], 1ull << i);
    }
  }
  __syncthreads();

  if (tid < NNODE)
    dinv_s[tid] = rsqrtf(1.0f + (float)__popcll(colmask[tid]));
  __syncthreads();

  for (int q = tid; q < NNODE * NOPS; q += 256)
    msg_s[q >> 3][q & 7] = dinv_s[q >> 3] * hproj[(size_t)b * NNODE * NOPS + q];
  __syncthreads();

  for (int q = tid; q < NNODE * NOPS; q += 256) {
    int j = q >> 3, o = q & 7;
    float acc = msg_s[j][o];                         // self loop
    unsigned long long m = colmask[j];
    while (m) { int i = __builtin_ctzll(m); acc += msg_s[i][o]; m &= m - 1; }
    emb_s[j][o] = dinv_s[j] * acc + bm[o];
  }
  __syncthreads();

  if (tid < NNODE) {
    const float* go = g_ops + ((size_t)b * NNODE + tid) * NOPS;
    int best = 0; float bv = emb_s[tid][0] + go[0];
    for (int o = 1; o < NOPS; ++o) {
      float v = emb_s[tid][o] + go[o];
      if (v > bv) { bv = v; best = o; }
    }
    float* op = out + ((size_t)b * NNODE + tid) * NOPS;
    for (int o = 0; o < NOPS; ++o) op[o] = (o == best) ? 1.0f : 0.0f;
  }
}

// ---------------- launcher ----------------
extern "C" void kernel_launch(void* const* d_in, const int* in_sizes, int n_in,
                              void* d_out, int out_size, void* d_ws, size_t ws_size,
                              hipStream_t stream) {
  const float* z  = (const float*)d_in[0];
  const float* ge = (const float*)d_in[1];
  const float* go = (const float*)d_in[2];
  const float* W1 = (const float*)d_in[3];
  const float* b1 = (const float*)d_in[4];
  const float* W2 = (const float*)d_in[5];
  const float* b2 = (const float*)d_in[6];
  const float* W3 = (const float*)d_in[7];
  const float* b3 = (const float*)d_in[8];
  const float* We = (const float*)d_in[9];
  const float* be = (const float*)d_in[10];
  const float* Wn = (const float*)d_in[11];
  const float* bn = (const float*)d_in[12];
  const float* Wm = (const float*)d_in[13];
  const float* bm = (const float*)d_in[14];
  float* out = (float*)d_out;

  size_t off = 0;
  auto alloc = [&](size_t bytes) -> void* {
    void* p = (char*)d_ws + off;
    off += (bytes + 255) & ~(size_t)255;
    return p;
  };
  unsigned short* zb   = (unsigned short*)alloc((size_t)BATCH * ZDIM * 2);
  unsigned short* W1t  = (unsigned short*)alloc((size_t)ZDIM * H1 * 2);   // [H1][ZDIM]
  unsigned short* W2t  = (unsigned short*)alloc((size_t)H1 * H2 * 2);     // [H2][H1]
  unsigned short* W3t  = (unsigned short*)alloc((size_t)H2 * H3 * 2);     // [H3][H2]
  unsigned short* Wdt  = (unsigned short*)alloc((size_t)H3 * NN2 * 2);    // [NN2][H3]
  float*          bd   = (float*)alloc((size_t)NN2 * 4);
  float*          Wftm = (float*)alloc((size_t)H3 * NNODE * NOPS * 4);    // [H3][512] f32 tmp
  unsigned short* Wft  = (unsigned short*)alloc((size_t)H3 * NNODE * NOPS * 2); // [512][H3]
  float*          bfv  = (float*)alloc((size_t)NNODE * NOPS * 4);
  unsigned short* h1   = (unsigned short*)alloc((size_t)BATCH * H1 * 2);
  unsigned short* h2   = (unsigned short*)alloc((size_t)BATCH * H2 * 2);
  unsigned short* h3   = (unsigned short*)alloc((size_t)BATCH * H3 * 2);
  float*          dsc  = (float*)alloc((size_t)BATCH * NN2 * 4);
  float*          hpj  = (float*)alloc((size_t)BATCH * NNODE * NOPS * 4);

  // activations: z stays row-major bf16
  k_cvt_bf16<<<(BATCH * ZDIM) / 256, 256, 0, stream>>>(z, zb, (size_t)BATCH * ZDIM);

  // weights: transpose + convert (dst[c][r] layouts), block (32,8), grid (C/32, R/32)
  dim3 tb(32, 8);
  k_transpose_cvt<false><<<dim3(H1 / 32, ZDIM / 32), tb, 0, stream>>>(W1, H1, W1t, ZDIM, H1);
  k_transpose_cvt<false><<<dim3(H2 / 32, H1 / 32),   tb, 0, stream>>>(W2, H2, W2t, H1, H2);
  k_transpose_cvt<false><<<dim3(H3 / 32, H2 / 32),   tb, 0, stream>>>(W3, H3, W3t, H2, H3);
  k_transpose_cvt<true ><<<dim3(NN2 / 32, H3 / 32),  tb, 0, stream>>>(We, 2 * NN2, Wdt, H3, NN2);
  {
    size_t nf = (size_t)H3 * NNODE * NOPS;
    k_prep_wf<<<(unsigned)((nf + 255) / 256), 256, 0, stream>>>(Wn, Wm, Wftm);
    k_transpose_cvt<false><<<dim3((NNODE * NOPS) / 32, H3 / 32), tb, 0, stream>>>(
        Wftm, NNODE * NOPS, Wft, H3, NNODE * NOPS);
    k_prep_bd<<<NN2 / 256, 256, 0, stream>>>(be, bd);
    k_prep_bf<<<2, 256, 0, stream>>>(bn, Wm, bfv);
  }

  // GEMM chain (grid: x = N/128, y = M/128)
  k_gemm_bf16<true ><<<dim3(H1 / 128, BATCH / 128), 256, 0, stream>>>(zb, W1t, b1, h1, H1, ZDIM);
  k_gemm_bf16<true ><<<dim3(H2 / 128, BATCH / 128), 256, 0, stream>>>(h1, W2t, b2, h2, H2, H1);
  k_gemm_bf16<true ><<<dim3(H3 / 128, BATCH / 128), 256, 0, stream>>>(h2, W3t, b3, h3, H3, H2);
  k_gemm_bf16<false><<<dim3(NN2 / 128, BATCH / 128), 256, 0, stream>>>(h3, Wdt, bd, dsc, NN2, H3);
  k_gemm_bf16<false><<<dim3((NNODE * NOPS) / 128, BATCH / 128), 256, 0, stream>>>(
      h3, Wft, bfv, hpj, NNODE * NOPS, H3);

  k_graph<<<BATCH, 256, 0, stream>>>(dsc, hpj, ge, go, bm, out);
}
